// DeformConvBlock_40303973106011
// MI455X (gfx1250) — matrix operand order repaired and checked by reading, not compiled, not run on hardware
//
#include <hip/hip_runtime.h>
#include <hip/hip_bf16.h>

typedef __attribute__((ext_vector_type(16))) _Float16 v16h;
typedef __attribute__((ext_vector_type(8)))  float    v8f;

#define Bn   8
#define Cc   64
#define Oo   64
#define Hh   128
#define Ww   128
#define HW   (Hh*Ww)          // 16384 = 2^14
#define KKd  576              // C * 9
#define NOFF 18

// workspace layout (bytes), ws base is 256B aligned
#define WS_WDEF 0                          // 64*576 f16 = 73728 B   ([o][k'], permuted K)
#define WS_WOFF 73728                      // 32*576 f16 = 36864 B   (rows 18..31 zero)
#define WS_X16  110592                     // NHWC f16 x: 8*128*128*64*2 = 16777216 B
#define WS_OFFS (110592 + 16777216)        // 8*18*128*128 f32 = 9437184 B
#define WS_STAT (110592 + 16777216 + 9437184) // 64 * {scale, shift} f32

// chunk-local K index l (0..31) -> channel offset: identity (contiguous order).
// Lanes 0-15 own c 0..15 (in order) and lanes 16-31 own c 16..31 (in order)
// in both A and B fragments => contiguous 32B NHWC loads.
__device__ __forceinline__ int kswap(int l) {
    return l;
}

__device__ __forceinline__ v16h splat16(float f) {
    _Float16 h = (_Float16)f;
    v16h v;
    #pragma unroll
    for (int i = 0; i < 16; ++i) v[i] = h;
    return v;
}

// ---------------------------------------------------------------------------
// x (NCHW f32) -> x16 (NHWC f16) via LDS tile: coalesced reads along W,
// coalesced f16 writes along C. Tile = 32 spatial x 64 channels per block.
// ---------------------------------------------------------------------------
__global__ __launch_bounds__(256) void x_to_nhwc(const float* __restrict__ x,
                                                 unsigned char* __restrict__ ws) {
    __shared__ float tile[Cc][33];
    _Float16* x16 = (_Float16*)(ws + WS_X16);
    int S0  = blockIdx.x * 32;          // global spatial index b*HW + s, 4096 blocks
    int b   = S0 >> 14;
    int sp0 = S0 & (HW - 1);
    int t   = threadIdx.x;
    #pragma unroll
    for (int p = 0; p < 8; ++p) {
        int c = p * 8 + (t >> 5);
        int s = t & 31;
        tile[c][s] = x[(b * Cc + c) * HW + sp0 + s];
    }
    __syncthreads();
    #pragma unroll
    for (int p = 0; p < 8; ++p) {
        int s = p * 4 + (t >> 6);
        int c = t & 63;
        x16[(S0 + s) * Cc + c] = (_Float16)tile[c][s];
    }
}

// ---------------------------------------------------------------------------
// Weights fp32 -> f16 with K' = kc*32 + l, where kk = kc>>1,
// c = (kc&1)*32 + kswap(l). B rows then match the permuted A fragments.
// ---------------------------------------------------------------------------
__global__ __launch_bounds__(256) void cvt_weights(const float* __restrict__ wdef,
                                                   const float* __restrict__ woff,
                                                   unsigned char* __restrict__ ws) {
    _Float16* wd = (_Float16*)(ws + WS_WDEF);
    _Float16* wo = (_Float16*)(ws + WS_WOFF);
    int i = blockIdx.x * 256 + threadIdx.x;     // grid covers 64*576 = 36864
    if (i < Oo * KKd) {
        int o  = i / KKd;
        int kp = i - o * KKd;
        int kc = kp >> 5;
        int l  = kp & 31;
        int kk = kc >> 1;
        int c  = (kc & 1) * 32 + kswap(l);
        wd[i] = (_Float16)wdef[(o * Cc + c) * 9 + kk];
    }
    if (i < 32 * KKd) {
        int n  = i / KKd;
        int kp = i - n * KKd;
        int kc = kp >> 5;
        int l  = kp & 31;
        int kk = kc >> 1;
        int c  = (kc & 1) * 32 + kswap(l);
        wo[i] = (n < NOFF) ? (_Float16)woff[(n * Cc + c) * 9 + kk] : (_Float16)0.0f;
    }
}

// ---------------------------------------------------------------------------
// Offset conv: implicit GEMM on v_wmma_f32_16x16x32_f16. Per wave: 16 spatial
// positions (M), 32 output channels (N, 18 used), K = 576 in 18 chunks.
// A fragment = single contiguous v16h NHWC load per chunk-half.
// ---------------------------------------------------------------------------
__global__ __launch_bounds__(256) void offset_conv(const float* __restrict__ b_off,
                                                   unsigned char* __restrict__ ws) {
    const v16h* xv = (const v16h*)(ws + WS_X16);
    const v16h* wov = (const v16h*)(ws + WS_WOFF);
    float* offs = (float*)(ws + WS_OFFS);

    int wave = blockIdx.x * (blockDim.x >> 5) + (threadIdx.x >> 5);
    int lane = threadIdx.x & 31;
    int b  = wave >> 10;
    int r  = wave & 1023;
    int hh = r >> 3;
    int w0 = (r & 7) << 4;
    int m  = lane & 15;               // A row (spatial) / D column (channel)
    int w  = w0 + m;
    int chalf = (lane < 16) ? 0 : 16; // lane's channel run within a chunk
    int klo   = (lane < 16) ? 0 : 16; // B-fragment K half

    v8f acc0 = {}, acc1 = {};

    for (int kk = 0; kk < 9; ++kk) {
        int ky = kk / 3, kx = kk - 3 * (kk / 3);
        int yy = hh - 1 + ky;
        int xx = w  - 1 + kx;
        bool inb = ((unsigned)yy < Hh) && ((unsigned)xx < Ww);
        int Sxy = b * HW + yy * Ww + xx;        // NHWC spatial index
        #pragma unroll
        for (int half = 0; half < 2; ++half) {
            int cstart = half * 32 + chalf;
            v16h a = {};
            if (inb) a = xv[Sxy * 4 + (cstart >> 4)];
            int kc = kk * 2 + half;
            v16h b0 = wov[( m       * KKd + kc * 32 + klo) >> 4];
            v16h b1 = wov[((m + 16) * KKd + kc * 32 + klo) >> 4];
            acc0 = __builtin_amdgcn_wmma_f32_16x16x32_f16(false, a, false, b0, (short)0, acc0, false, false);
            acc1 = __builtin_amdgcn_wmma_f32_16x16x32_f16(false, a, false, b1, (short)0, acc1, false, false);
        }
    }

    int mo = (lane < 16) ? 0 : 8;     // D row offset per half-wave
    #pragma unroll
    for (int rI = 0; rI < 8; ++rI) {
        int wsp = w0 + rI + mo;
        offs[((b * NOFF + m) * Hh + hh) * Ww + wsp] = acc0[rI] + b_off[m];
        int n1 = m + 16;
        if (n1 < NOFF)
            offs[((b * NOFF + n1) * Hh + hh) * Ww + wsp] = acc1[rI] + b_off[n1];
    }
}

// ---------------------------------------------------------------------------
// Deformable GEMM: per wave M = 16 positions, N = 64 (4 WMMA tiles), K = 576.
// Bilinear sample = 4 contiguous v16h NHWC loads + packed-f16 blend per chunk.
// Epilogue: LDS transpose (16 pos x 64 ch per wave) -> coalesced b64 stores.
// ---------------------------------------------------------------------------
__global__ __launch_bounds__(256) void deform_gemm(const float* __restrict__ b_def,
                                                   float* __restrict__ out,
                                                   const unsigned char* __restrict__ ws) {
    __shared__ float ot[8][Oo][16];   // 32 KB: per-wave 16x64 f32 output tile
    const v16h* xv  = (const v16h*)(ws + WS_X16);
    const v16h* wdv = (const v16h*)(ws + WS_WDEF);
    const float* offs = (const float*)(ws + WS_OFFS);

    int wv   = threadIdx.x >> 5;
    int wave = blockIdx.x * (blockDim.x >> 5) + wv;
    int lane = threadIdx.x & 31;
    int b  = wave >> 10;
    int r  = wave & 1023;
    int hh = r >> 3;
    int w0 = (r & 7) << 4;
    int m  = lane & 15;
    int w  = w0 + m;
    int chalf = (lane < 16) ? 0 : 16;
    int klo   = (lane < 16) ? 0 : 16;

    v8f acc[4] = {};

    for (int kk = 0; kk < 9; ++kk) {
        int ky = kk / 3, kx = kk - 3 * (kk / 3);
        float dyv = offs[((b * NOFF + 2 * kk    ) * Hh + hh) * Ww + w];
        float dxv = offs[((b * NOFF + 2 * kk + 1) * Hh + hh) * Ww + w];
        float py = dyv + (float)(hh - 1 + ky);
        float px = dxv + (float)(w  - 1 + kx);
        float y0f = floorf(py), x0f = floorf(px);
        float fy = py - y0f, fx = px - x0f;
        int y0 = (int)y0f, x0 = (int)x0f;
        int y1 = y0 + 1,   x1 = x0 + 1;
        float vy0 = ((unsigned)y0 < Hh) ? 1.0f : 0.0f;
        float vy1 = ((unsigned)y1 < Hh) ? 1.0f : 0.0f;
        float vx0 = ((unsigned)x0 < Ww) ? 1.0f : 0.0f;
        float vx1 = ((unsigned)x1 < Ww) ? 1.0f : 0.0f;
        int y0c = y0 < 0 ? 0 : (y0 > Hh - 1 ? Hh - 1 : y0);
        int y1c = y1 < 0 ? 0 : (y1 > Hh - 1 ? Hh - 1 : y1);
        int x0c = x0 < 0 ? 0 : (x0 > Ww - 1 ? Ww - 1 : x0);
        int x1c = x1 < 0 ? 0 : (x1 > Ww - 1 ? Ww - 1 : x1);
        v16h W00 = splat16((1.0f - fy) * (1.0f - fx) * vy0 * vx0);
        v16h W01 = splat16((1.0f - fy) * fx          * vy0 * vx1);
        v16h W10 = splat16(fy          * (1.0f - fx) * vy1 * vx0);
        v16h W11 = splat16(fy          * fx          * vy1 * vx1);
        // NHWC spatial bases (v16h units = spatial*4)
        int s00 = (b * HW + y0c * Ww + x0c) * 4;
        int s01 = (b * HW + y0c * Ww + x1c) * 4;
        int s10 = (b * HW + y1c * Ww + x0c) * 4;
        int s11 = (b * HW + y1c * Ww + x1c) * 4;

        #pragma unroll
        for (int half = 0; half < 2; ++half) {
            int cs4 = (half * 32 + chalf) >> 4;
            v16h v00 = xv[s00 + cs4];
            v16h v01 = xv[s01 + cs4];
            v16h v10 = xv[s10 + cs4];
            v16h v11 = xv[s11 + cs4];
            v16h a = W00 * v00 + W01 * v01 + W10 * v10 + W11 * v11;  // v_pk_fma_f16
            int kc = kk * 2 + half;
            #pragma unroll
            for (int nt = 0; nt < 4; ++nt) {
                v16h bf = wdv[((nt * 16 + m) * KKd + kc * 32 + klo) >> 4];
                acc[nt] = __builtin_amdgcn_wmma_f32_16x16x32_f16(false, a, false, bf, (short)0, acc[nt], false, false);
            }
        }
    }

    // Epilogue: transpose D fragments through LDS so stores are channel-major.
    int mo = (lane < 16) ? 0 : 8;
    #pragma unroll
    for (int nt = 0; nt < 4; ++nt) {
        #pragma unroll
        for (int rI = 0; rI < 8; ++rI)
            ot[wv][nt * 16 + m][mo + rI] = acc[nt][rI];
    }
    __syncthreads();
    int n4 = lane >> 3;               // 4 channels per iteration
    int wl = (lane & 7) * 2;          // 2 consecutive w per lane -> b64 stores
    #pragma unroll
    for (int it = 0; it < 16; ++it) {
        int n = it * 4 + n4;
        float bias = b_def[n];
        float2 v;
        v.x = ot[wv][n][wl]     + bias;
        v.y = ot[wv][n][wl + 1] + bias;
        *(float2*)(out + ((b * Oo + n) * Hh + hh) * Ww + w0 + wl) = v;
    }
}

// ---------------------------------------------------------------------------
// BatchNorm: per-channel sum/sumsq reduction -> fused scale/shift, then apply.
// ---------------------------------------------------------------------------
__global__ __launch_bounds__(256) void bn_reduce(const float* __restrict__ out,
                                                 const float* __restrict__ gamma,
                                                 const float* __restrict__ beta,
                                                 unsigned char* __restrict__ ws) {
    __shared__ float ssum[256], ssq[256];
    int o = blockIdx.x;
    float s = 0.0f, q = 0.0f;
    for (int b = 0; b < Bn; ++b) {
        const float* p = out + (size_t)(b * Oo + o) * HW;
        for (int i = threadIdx.x; i < HW; i += 256) {
            float v = p[i];
            s += v; q += v * v;
        }
    }
    ssum[threadIdx.x] = s; ssq[threadIdx.x] = q;
    __syncthreads();
    for (int st = 128; st > 0; st >>= 1) {
        if ((int)threadIdx.x < st) {
            ssum[threadIdx.x] += ssum[threadIdx.x + st];
            ssq[threadIdx.x]  += ssq[threadIdx.x + st];
        }
        __syncthreads();
    }
    if (threadIdx.x == 0) {
        float n    = (float)(Bn * HW);
        float mean = ssum[0] / n;
        float var  = ssq[0] / n - mean * mean;
        float inv  = rsqrtf(var + 1e-5f);
        float sc   = inv * gamma[o];
        float* st2 = (float*)(ws + WS_STAT);
        st2[2 * o]     = sc;
        st2[2 * o + 1] = beta[o] - mean * sc;
    }
}

__global__ __launch_bounds__(256) void bn_apply(float* __restrict__ out,
                                                const unsigned char* __restrict__ ws) {
    const float* st = (const float*)(ws + WS_STAT);
    int i = blockIdx.x * 256 + threadIdx.x;
    if (i >= Bn * Oo * HW) return;
    int o = (i >> 14) & 63;           // (i / HW) % Oo, HW = 2^14
    out[i] = out[i] * st[2 * o] + st[2 * o + 1];
}

// ---------------------------------------------------------------------------
extern "C" void kernel_launch(void* const* d_in, const int* in_sizes, int n_in,
                              void* d_out, int out_size, void* d_ws, size_t ws_size,
                              hipStream_t stream) {
    const float* x     = (const float*)d_in[0];
    const float* w_off = (const float*)d_in[1];
    const float* b_off = (const float*)d_in[2];
    const float* w_def = (const float*)d_in[3];
    const float* b_def = (const float*)d_in[4];
    const float* gamma = (const float*)d_in[5];
    const float* beta  = (const float*)d_in[6];
    float* out = (float*)d_out;
    unsigned char* ws = (unsigned char*)d_ws;

    x_to_nhwc  <<<4096, 256, 0, stream>>>(x, ws);
    cvt_weights<<<144, 256, 0, stream>>>(w_def, w_off, ws);
    offset_conv<<<1024, 256, 0, stream>>>(b_off, ws);          // 8192 waves, 1 tile each
    deform_gemm<<<1024, 256, 0, stream>>>(b_def, out, ws);     // 8192 waves, 1 tile each
    bn_reduce  <<<Oo, 256, 0, stream>>>(out, gamma, beta, ws);
    int n = Bn * Oo * HW;
    bn_apply   <<<(n + 255) / 256, 256, 0, stream>>>(out, ws);
}
